// Flux3_13932873909032
// MI455X (gfx1250) — matrix operand order repaired and checked by reading, compiled, not run on hardware
//
#include <hip/hip_runtime.h>
#include <hip/hip_bf16.h>

#define LTXT 256
#define LIMG 2048
#define LSEQ (LTXT + LIMG)
#define DMODEL 2048
#define NH 16
#define DHEAD 128
#define DMLP 8192
#define EPSF 1e-6f

typedef unsigned short u16;
typedef __bf16 bf16_t;
typedef bf16_t bf16x16 __attribute__((ext_vector_type(16)));
typedef float v8f __attribute__((ext_vector_type(8)));
typedef int v4i __attribute__((vector_size(16)));

union Frag { bf16x16 v; uint4 q[2]; };

// gfx1250 async global->LDS copy path (ASYNCcnt), with sync fallback.
#if defined(__gfx1250__) && __has_builtin(__builtin_amdgcn_global_load_async_to_lds_b128) && __has_builtin(__builtin_amdgcn_s_wait_asynccnt)
#define USE_ASYNC_COPY 1
#else
#define USE_ASYNC_COPY 0
#endif

template <int NB>
__device__ __forceinline__ void stage_copy(u16* lds, const u16* g) {
#if USE_ASYNC_COPY
#pragma unroll
  for (int i = 0; i < NB; ++i)
    __builtin_amdgcn_global_load_async_to_lds_b128(
        (v4i*)(void*)(g + i * 8), (v4i*)(void*)(lds + i * 8), 0, 0);
#else
  uint4 tmp[NB];
#pragma unroll
  for (int i = 0; i < NB; ++i) tmp[i] = ((const uint4*)g)[i];
#pragma unroll
  for (int i = 0; i < NB; ++i) ((uint4*)lds)[i] = tmp[i];
#endif
}

__device__ __forceinline__ void stage_wait() {
#if USE_ASYNC_COPY
  __builtin_amdgcn_s_wait_asynccnt(0);
#endif
}

__device__ __forceinline__ u16 f2bf(float f) {
  unsigned u = __float_as_uint(f);
  unsigned r = u + 0x7fffu + ((u >> 16) & 1u);
  return (u16)(r >> 16);
}

__device__ __forceinline__ v8f wmma_bf16(bf16x16 a, bf16x16 b, v8f c) {
  // D = A(16x32 bf16) * B(32x16 bf16) + C(16x16 f32)
  return __builtin_amdgcn_wmma_f32_16x16x32_bf16(false, a, false, b, (short)0, c,
                                                 false, false);
}

__device__ __forceinline__ float gelu_tanh(float x) {
  float x3 = x * x * x;
  return 0.5f * x * (1.0f + tanhf(0.7978845608028654f * (x + 0.044715f * x3)));
}

__device__ __forceinline__ float rmax16(float v) {
  v = fmaxf(v, __shfl_xor(v, 1, 32));
  v = fmaxf(v, __shfl_xor(v, 2, 32));
  v = fmaxf(v, __shfl_xor(v, 4, 32));
  v = fmaxf(v, __shfl_xor(v, 8, 32));
  return v;
}
__device__ __forceinline__ float rsum16(float v) {
  v += __shfl_xor(v, 1, 32);
  v += __shfl_xor(v, 2, 32);
  v += __shfl_xor(v, 4, 32);
  v += __shfl_xor(v, 8, 32);
  return v;
}

// ---------------------------------------------------------------------------
// Weight cast f32[K][N] -> bf16 transposed [N][K]
// ---------------------------------------------------------------------------
__global__ __launch_bounds__(256) void cast_transpose_kernel(
    const float* __restrict__ in, u16* __restrict__ out, int K, int N) {
  __shared__ float tile[32][33];
  const int n0 = blockIdx.x * 32;
  const int k0 = blockIdx.y * 32;
  const int tx = threadIdx.x & 31;
  const int ty = threadIdx.x >> 5;
#pragma unroll
  for (int i = 0; i < 4; ++i) {
    int k = ty + i * 8;
    tile[k][tx] = in[(size_t)(k0 + k) * N + n0 + tx];
  }
  __syncthreads();
#pragma unroll
  for (int i = 0; i < 4; ++i) {
    int n = ty + i * 8;
    out[(size_t)(n0 + n) * K + k0 + tx] = f2bf(tile[tx][n]);
  }
}

// ---------------------------------------------------------------------------
// mod = silu(vec) @ W + b   (vec 2048, out 12288)
// ---------------------------------------------------------------------------
__global__ __launch_bounds__(256) void mod_kernel(
    const float* __restrict__ vec, const float* __restrict__ w,
    const float* __restrict__ b, float* __restrict__ out) {
  int j = blockIdx.x * 256 + threadIdx.x;
  float acc = 0.f;
  for (int k = 0; k < DMODEL; ++k) {
    float v = vec[k];
    float sv = v / (1.0f + __expf(-v));
    acc += sv * w[(size_t)k * (6 * DMODEL) + j];
  }
  out[j] = acc + b[j];
}

// ---------------------------------------------------------------------------
// xm = (1+sc)*LN(x) + sh, cast to bf16. One block per row.
// ---------------------------------------------------------------------------
__global__ __launch_bounds__(256) void ln_mod_kernel(
    const float* __restrict__ x, const float* __restrict__ sh,
    const float* __restrict__ sc, u16* __restrict__ out) {
  const int row = blockIdx.x;
  const int tid = threadIdx.x;
  const float* xr = x + (size_t)row * DMODEL;
  float vals[8];
  float s = 0.f, s2 = 0.f;
#pragma unroll
  for (int i = 0; i < 8; ++i) {
    float v = xr[tid + i * 256];
    vals[i] = v;
    s += v;
    s2 += v * v;
  }
#pragma unroll
  for (int m = 1; m < 32; m <<= 1) {
    s += __shfl_xor(s, m, 32);
    s2 += __shfl_xor(s2, m, 32);
  }
  __shared__ float rs[8], rs2[8];
  int wid = tid >> 5, lane = tid & 31;
  if (lane == 0) { rs[wid] = s; rs2[wid] = s2; }
  __syncthreads();
  s = 0.f; s2 = 0.f;
#pragma unroll
  for (int i = 0; i < 8; ++i) { s += rs[i]; s2 += rs2[i]; }
  float mean = s * (1.0f / DMODEL);
  float var = s2 * (1.0f / DMODEL) - mean * mean;
  float inv = rsqrtf(var + EPSF);
  u16* orow = out + (size_t)row * DMODEL;
#pragma unroll
  for (int i = 0; i < 8; ++i) {
    int d = tid + i * 256;
    float nv = (vals[i] - mean) * inv;
    orow[d] = f2bf((1.0f + sc[d]) * nv + sh[d]);
  }
}

// ---------------------------------------------------------------------------
// Generic bf16 WMMA GEMM.  A [M x K] bf16 row-major, Wt [N x K] bf16.
// Block = 128x128 tile, 8 waves (2x4), wave = 64x32.  BK = 32.
// Double-buffered LDS, async global->LDS staging.
// mode 0: outf = acc + bias (f32)
// mode 1: outb = bf16(gelu(acc + bias))
// mode 2: resid += gate * (acc + bias)
// ---------------------------------------------------------------------------
__global__ __launch_bounds__(256) void gemm_bf16_kernel(
    const u16* __restrict__ A, const u16* __restrict__ Wt,
    const float* __restrict__ bias, int K, int mode,
    float* __restrict__ outf, int ldof,
    u16* __restrict__ outb, int ldob,
    const float* __restrict__ gate, float* __restrict__ resid, int ldr) {
  __shared__ u16 lA[2][128 * 32];
  __shared__ u16 lB[2][128 * 32];
  const int m0 = blockIdx.y * 128;
  const int n0 = blockIdx.x * 128;
  const int tid = threadIdx.x;
  const int wid = tid >> 5, lane = tid & 31;
  const int wrow = wid >> 2, wcol = wid & 3;
  const int hf = lane >> 4, l15 = lane & 15;

  v8f acc[4][2];
#pragma unroll
  for (int mi = 0; mi < 4; ++mi)
#pragma unroll
    for (int ni = 0; ni < 2; ++ni)
#pragma unroll
      for (int r = 0; r < 8; ++r) acc[mi][ni][r] = 0.0f;

  const int trow = tid >> 1;
  const int tcol = (tid & 1) << 4;
  const u16* gA = A + (size_t)(m0 + trow) * K + tcol;
  const u16* gB = Wt + (size_t)(n0 + trow) * K + tcol;
  const int lofs = trow * 32 + tcol;

  // prologue: stage k-slice 0 into buffer 0
  stage_copy<2>(&lA[0][lofs], gA);
  stage_copy<2>(&lB[0][lofs], gB);

  int cur = 0;
  for (int k0 = 0; k0 < K; k0 += 32) {
    stage_wait();      // my async copies for buf[cur] landed
    __syncthreads();   // everyone's copies landed; buf[cur^1] free to overwrite
    if (k0 + 32 < K) {
      stage_copy<2>(&lA[cur ^ 1][lofs], gA + k0 + 32);
      stage_copy<2>(&lB[cur ^ 1][lofs], gB + k0 + 32);
      __builtin_prefetch(gA + k0 + 64, 0, 1);
      __builtin_prefetch(gB + k0 + 64, 0, 1);
    }

    Frag af[4], bfw[2];
#pragma unroll
    for (int mi = 0; mi < 4; ++mi) {
      int row = wrow * 64 + mi * 16 + l15;
      int kb = hf * 16;  // lanes 0-15: K 0..15; lanes 16-31: K 16..31
      af[mi].q[0] = *(const uint4*)&lA[cur][row * 32 + kb];
      af[mi].q[1] = *(const uint4*)&lA[cur][row * 32 + kb + 8];
    }
#pragma unroll
    for (int ni = 0; ni < 2; ++ni) {
      int n = wcol * 32 + ni * 16 + l15;
      int kb = hf * 16;  // lanes 0-15: K 0..15; lanes 16-31: K 16..31
      bfw[ni].q[0] = *(const uint4*)&lB[cur][n * 32 + kb];
      bfw[ni].q[1] = *(const uint4*)&lB[cur][n * 32 + kb + 8];
    }
#pragma unroll
    for (int mi = 0; mi < 4; ++mi)
#pragma unroll
      for (int ni = 0; ni < 2; ++ni)
        acc[mi][ni] = wmma_bf16(af[mi].v, bfw[ni].v, acc[mi][ni]);
    cur ^= 1;
  }

#pragma unroll
  for (int mi = 0; mi < 4; ++mi) {
#pragma unroll
    for (int ni = 0; ni < 2; ++ni) {
      int gn = n0 + wcol * 32 + ni * 16 + l15;
      int gmb = m0 + wrow * 64 + mi * 16 + 8 * hf;
      float bc = bias[gn];
      float gc = (mode == 2) ? gate[gn] : 0.0f;
#pragma unroll
      for (int r = 0; r < 8; ++r) {
        int gm = gmb + r;
        float val = acc[mi][ni][r] + bc;
        if (mode == 0) {
          outf[(size_t)gm * ldof + gn] = val;
        } else if (mode == 1) {
          outb[(size_t)gm * ldob + gn] = f2bf(gelu_tanh(val));
        } else {
          resid[(size_t)gm * ldr + gn] += gc * val;
        }
      }
    }
  }
}

// ---------------------------------------------------------------------------
// RMS-norm q/k, RoPE, pack to head-major bf16.  Block = 128 threads = 1 (u,h).
// q,k -> [h][u][dh] ;  v -> transposed [h][dh][u]
// ---------------------------------------------------------------------------
__global__ __launch_bounds__(128) void qk_rope_pack_kernel(
    const float* __restrict__ qkvf, const float* __restrict__ pe,
    const float* __restrict__ qn_img, const float* __restrict__ kn_img,
    const float* __restrict__ qn_txt, const float* __restrict__ kn_txt,
    u16* __restrict__ qb, u16* __restrict__ kg, u16* __restrict__ vtb) {
  const int u = blockIdx.x;
  const int h = blockIdx.y;
  const int dh = threadIdx.x;
  const int wid = dh >> 5, lane = dh & 31;
  const float* qn = (u < LTXT) ? qn_txt : qn_img;
  const float* kn = (u < LTXT) ? kn_txt : kn_img;
  const float* row = qkvf + (size_t)u * (3 * DMODEL);
  float q = row[(0 * NH + h) * DHEAD + dh];
  float k = row[(1 * NH + h) * DHEAD + dh];
  float v = row[(2 * NH + h) * DHEAD + dh];
  float q2 = q * q, k2 = k * k;
#pragma unroll
  for (int m = 1; m < 32; m <<= 1) {
    q2 += __shfl_xor(q2, m, 32);
    k2 += __shfl_xor(k2, m, 32);
  }
  __shared__ float rq[4], rk[4];
  if (lane == 0) { rq[wid] = q2; rk[wid] = k2; }
  __syncthreads();
  float qs = rq[0] + rq[1] + rq[2] + rq[3];
  float ks = rk[0] + rk[1] + rk[2] + rk[3];
  q = q * rsqrtf(qs * (1.0f / DHEAD) + EPSF) * qn[dh];
  k = k * rsqrtf(ks * (1.0f / DHEAD) + EPSF) * kn[dh];
  // RoPE: pe[l][j][a][{0,1}] ; out_a = pe0*x_even + pe1*x_odd
  int j = dh >> 1, a = dh & 1;
  const float* pp = pe + (((size_t)u * (DHEAD / 2) + j) * 2 + a) * 2;
  float c = pp[0], sn = pp[1];
  float qp = __shfl_xor(q, 1, 32);
  float kp = __shfl_xor(k, 1, 32);
  float q0 = a ? qp : q, q1 = a ? q : qp;
  float k0 = a ? kp : k, k1 = a ? k : kp;
  size_t oq = ((size_t)h * LSEQ + u) * DHEAD + dh;
  qb[oq] = f2bf(c * q0 + sn * q1);
  kg[oq] = f2bf(c * k0 + sn * k1);
  vtb[((size_t)h * DHEAD + dh) * LSEQ + u] = f2bf(v);
}

// ---------------------------------------------------------------------------
// Flash attention.  Block = (128 q-rows, 1 head), 4 waves, wave = 32 q-rows.
// Key chunks of 64.  scores & pv on bf16 WMMA, online softmax in f32.
// ---------------------------------------------------------------------------
__global__ __launch_bounds__(128) void attn_kernel(
    const u16* __restrict__ qb, const u16* __restrict__ kg,
    const u16* __restrict__ vtb, u16* __restrict__ attnb) {
  const int h = blockIdx.y;
  const int q0 = blockIdx.x * 128;
  const int tid = threadIdx.x;
  const int wid = tid >> 5, lane = tid & 31;
  const int hf = lane >> 4, l15 = lane & 15;

  __shared__ u16 lK[64 * 128];    // [key][dh]
  __shared__ u16 lVt[128 * 64];   // [dh][key]
  __shared__ u16 lP[4][32 * 64];  // per-wave probs [row][key]

  const size_t hbase = (size_t)h * LSEQ * DHEAD;
  const float scale = 0.08838834764831843f;  // 1/sqrt(128)

  // Q fragments: 32 rows x 128 dh per wave, loaded once.
  Frag aq[2][4];
#pragma unroll
  for (int mi = 0; mi < 2; ++mi) {
    int r = q0 + wid * 32 + mi * 16 + l15;
    const u16* src = qb + hbase + (size_t)r * DHEAD;
#pragma unroll
    for (int kk = 0; kk < 4; ++kk) {
      int kbb = kk * 32 + hf * 16;
      aq[mi][kk].q[0] = *(const uint4*)(src + kbb);
      aq[mi][kk].q[1] = *(const uint4*)(src + kbb + 8);
    }
  }

  v8f o[2][8];
  float mstat[2][8], lstat[2][8];
#pragma unroll
  for (int mi = 0; mi < 2; ++mi)
#pragma unroll
    for (int r = 0; r < 8; ++r) {
      mstat[mi][r] = -1e30f;
      lstat[mi][r] = 0.f;
    }
#pragma unroll
  for (int mi = 0; mi < 2; ++mi)
#pragma unroll
    for (int nj = 0; nj < 8; ++nj)
#pragma unroll
      for (int r = 0; r < 8; ++r) o[mi][nj][r] = 0.f;

  const int krow = tid >> 1, kcb = (tid & 1) * 64;

  for (int kc = 0; kc < LSEQ; kc += 64) {
    __syncthreads();
    // stage K chunk [64 key][128 dh] and Vt chunk [128 dh][64 key] (async)
    stage_copy<8>(&lK[krow * 128 + kcb],
                  kg + hbase + (size_t)(kc + krow) * DHEAD + kcb);
    stage_copy<8>(&lVt[tid * 64], vtb + (size_t)(h * DHEAD + tid) * LSEQ + kc);
    stage_wait();
    __syncthreads();

    // ---- scores: S = Q @ K^T  (32 q-rows x 64 keys per wave) ----
    v8f sacc[2][4];
#pragma unroll
    for (int mi = 0; mi < 2; ++mi)
#pragma unroll
      for (int ni = 0; ni < 4; ++ni)
#pragma unroll
        for (int r = 0; r < 8; ++r) sacc[mi][ni][r] = 0.f;
#pragma unroll
    for (int kk = 0; kk < 4; ++kk) {
      Frag bk[4];
#pragma unroll
      for (int ni = 0; ni < 4; ++ni) {
        int n = ni * 16 + l15;        // key within chunk
        int kb2 = kk * 32 + hf * 16;  // dh slice
        bk[ni].q[0] = *(const uint4*)&lK[n * 128 + kb2];
        bk[ni].q[1] = *(const uint4*)&lK[n * 128 + kb2 + 8];
      }
#pragma unroll
      for (int mi = 0; mi < 2; ++mi)
#pragma unroll
        for (int ni = 0; ni < 4; ++ni)
          sacc[mi][ni] = wmma_bf16(aq[mi][kk].v, bk[ni].v, sacc[mi][ni]);
    }

    // ---- online softmax (row = vgpr slot + 8*laneHalf, cols across 16 lanes)
#pragma unroll
    for (int mi = 0; mi < 2; ++mi) {
#pragma unroll
      for (int r = 0; r < 8; ++r) {
        float pm = fmaxf(fmaxf(sacc[mi][0][r], sacc[mi][1][r]),
                         fmaxf(sacc[mi][2][r], sacc[mi][3][r])) * scale;
        pm = rmax16(pm);
        float mnew = fmaxf(mstat[mi][r], pm);
        float alpha = __expf(mstat[mi][r] - mnew);
        mstat[mi][r] = mnew;
        float psum = 0.f;
#pragma unroll
        for (int ni = 0; ni < 4; ++ni) {
          float p = __expf(sacc[mi][ni][r] * scale - mnew);
          sacc[mi][ni][r] = p;
          psum += p;
        }
        psum = rsum16(psum);
        lstat[mi][r] = lstat[mi][r] * alpha + psum;
#pragma unroll
        for (int nj = 0; nj < 8; ++nj) o[mi][nj][r] *= alpha;
      }
    }

    // ---- route probs C-layout -> A-layout via per-wave LDS ----
#pragma unroll
    for (int mi = 0; mi < 2; ++mi)
#pragma unroll
      for (int ni = 0; ni < 4; ++ni)
#pragma unroll
        for (int r = 0; r < 8; ++r) {
          int prow = mi * 16 + r + 8 * hf;
          lP[wid][prow * 64 + ni * 16 + l15] = f2bf(sacc[mi][ni][r]);
        }
    __syncthreads();

    // ---- O += P @ V  (32 rows x 128 dh per wave) ----
#pragma unroll
    for (int kkp = 0; kkp < 2; ++kkp) {
      Frag ap[2];
#pragma unroll
      for (int mi = 0; mi < 2; ++mi) {
        int arow = mi * 16 + l15;
        int kb3 = kkp * 32 + hf * 16;
        ap[mi].q[0] = *(const uint4*)&lP[wid][arow * 64 + kb3];
        ap[mi].q[1] = *(const uint4*)&lP[wid][arow * 64 + kb3 + 8];
      }
#pragma unroll
      for (int nj = 0; nj < 8; ++nj) {
        int n = nj * 16 + l15;  // dh
        int kb4 = kkp * 32 + hf * 16;
        Frag bv;
        bv.q[0] = *(const uint4*)&lVt[n * 64 + kb4];
        bv.q[1] = *(const uint4*)&lVt[n * 64 + kb4 + 8];
#pragma unroll
        for (int mi = 0; mi < 2; ++mi)
          o[mi][nj] = wmma_bf16(ap[mi].v, bv.v, o[mi][nj]);
      }
    }
  }

  // ---- normalize, de-interleave heads, store bf16 ----
#pragma unroll
  for (int mi = 0; mi < 2; ++mi)
#pragma unroll
    for (int nj = 0; nj < 8; ++nj)
#pragma unroll
      for (int r = 0; r < 8; ++r) {
        int row = q0 + wid * 32 + mi * 16 + r + 8 * hf;
        int col = h * DHEAD + nj * 16 + l15;
        float v = o[mi][nj][r] / lstat[mi][r];
        attnb[(size_t)row * DMODEL + col] = f2bf(v);
      }
}

// ---------------------------------------------------------------------------
extern "C" void kernel_launch(void* const* d_in, const int* in_sizes, int n_in,
                              void* d_out, int out_size, void* d_ws,
                              size_t ws_size, hipStream_t stream) {
  (void)in_sizes; (void)n_in; (void)out_size; (void)ws_size;

  const float* img = (const float*)d_in[0];
  const float* txt = (const float*)d_in[1];
  const float* vec = (const float*)d_in[2];
  const float* pe = (const float*)d_in[3];
  const float* img_mod_w = (const float*)d_in[4];
  const float* img_mod_b = (const float*)d_in[5];
  const float* img_qkv_w = (const float*)d_in[6];
  const float* img_qkv_b = (const float*)d_in[7];
  const float* img_qnorm = (const float*)d_in[8];
  const float* img_knorm = (const float*)d_in[9];
  const float* img_proj_w = (const float*)d_in[10];
  const float* img_proj_b = (const float*)d_in[11];
  const float* img_mlp0_w = (const float*)d_in[12];
  const float* img_mlp0_b = (const float*)d_in[13];
  const float* img_mlp1_w = (const float*)d_in[14];
  const float* img_mlp1_b = (const float*)d_in[15];
  const float* txt_mod_w = (const float*)d_in[16];
  const float* txt_mod_b = (const float*)d_in[17];
  const float* txt_qkv_w = (const float*)d_in[18];
  const float* txt_qkv_b = (const float*)d_in[19];
  const float* txt_qnorm = (const float*)d_in[20];
  const float* txt_knorm = (const float*)d_in[21];
  const float* txt_proj_w = (const float*)d_in[22];
  const float* txt_proj_b = (const float*)d_in[23];
  const float* txt_mlp0_w = (const float*)d_in[24];
  const float* txt_mlp0_b = (const float*)d_in[25];
  const float* txt_mlp1_w = (const float*)d_in[26];
  const float* txt_mlp1_b = (const float*)d_in[27];

  float* out = (float*)d_out;
  float* res_img = out;                          // [2048][2048]
  float* res_txt = out + (size_t)LIMG * DMODEL;  // [256][2048]

  size_t off = 0;
  auto alloc = [&](size_t bytes) -> void* {
    void* p = (char*)d_ws + off;
    off += (bytes + 255) & ~(size_t)255;
    return p;
  };
  u16* qkvT_img = (u16*)alloc((size_t)3 * DMODEL * DMODEL * 2);
  u16* qkvT_txt = (u16*)alloc((size_t)3 * DMODEL * DMODEL * 2);
  u16* projT_img = (u16*)alloc((size_t)DMODEL * DMODEL * 2);
  u16* projT_txt = (u16*)alloc((size_t)DMODEL * DMODEL * 2);
  u16* mlp0T_img = (u16*)alloc((size_t)DMLP * DMODEL * 2);
  u16* mlp0T_txt = (u16*)alloc((size_t)DMLP * DMODEL * 2);
  u16* mlp1T_img = (u16*)alloc((size_t)DMODEL * DMLP * 2);
  u16* mlp1T_txt = (u16*)alloc((size_t)DMODEL * DMLP * 2);
  float* mod_img = (float*)alloc((size_t)6 * DMODEL * 4);
  float* mod_txt = (float*)alloc((size_t)6 * DMODEL * 4);
  u16* xm = (u16*)alloc((size_t)LSEQ * DMODEL * 2);
  float* qkvf = (float*)alloc((size_t)LSEQ * 3 * DMODEL * 4);
  u16* qbuf = (u16*)alloc((size_t)NH * LSEQ * DHEAD * 2);
  u16* kbuf = (u16*)alloc((size_t)NH * LSEQ * DHEAD * 2);
  u16* vtb = (u16*)alloc((size_t)NH * DHEAD * LSEQ * 2);
  u16* attnb = (u16*)alloc((size_t)LSEQ * DMODEL * 2);
  u16* hid = (u16*)alloc((size_t)LSEQ * DMLP * 2);

  // residual init
  (void)hipMemcpyAsync(res_img, img, (size_t)LIMG * DMODEL * 4,
                       hipMemcpyDeviceToDevice, stream);
  (void)hipMemcpyAsync(res_txt, txt, (size_t)LTXT * DMODEL * 4,
                       hipMemcpyDeviceToDevice, stream);

  dim3 b256(256);
  // weight cast + transpose to bf16 [N][K]
  cast_transpose_kernel<<<dim3(6144 / 32, 2048 / 32), b256, 0, stream>>>(
      img_qkv_w, qkvT_img, 2048, 6144);
  cast_transpose_kernel<<<dim3(6144 / 32, 2048 / 32), b256, 0, stream>>>(
      txt_qkv_w, qkvT_txt, 2048, 6144);
  cast_transpose_kernel<<<dim3(2048 / 32, 2048 / 32), b256, 0, stream>>>(
      img_proj_w, projT_img, 2048, 2048);
  cast_transpose_kernel<<<dim3(2048 / 32, 2048 / 32), b256, 0, stream>>>(
      txt_proj_w, projT_txt, 2048, 2048);
  cast_transpose_kernel<<<dim3(8192 / 32, 2048 / 32), b256, 0, stream>>>(
      img_mlp0_w, mlp0T_img, 2048, 8192);
  cast_transpose_kernel<<<dim3(8192 / 32, 2048 / 32), b256, 0, stream>>>(
      txt_mlp0_w, mlp0T_txt, 2048, 8192);
  cast_transpose_kernel<<<dim3(2048 / 32, 8192 / 32), b256, 0, stream>>>(
      img_mlp1_w, mlp1T_img, 8192, 2048);
  cast_transpose_kernel<<<dim3(2048 / 32, 8192 / 32), b256, 0, stream>>>(
      txt_mlp1_w, mlp1T_txt, 8192, 2048);

  // modulation vectors
  mod_kernel<<<6 * DMODEL / 256, b256, 0, stream>>>(vec, img_mod_w, img_mod_b,
                                                    mod_img);
  mod_kernel<<<6 * DMODEL / 256, b256, 0, stream>>>(vec, txt_mod_w, txt_mod_b,
                                                    mod_txt);

  // LN1 + modulate (unified rows: txt first)
  ln_mod_kernel<<<LTXT, b256, 0, stream>>>(res_txt, mod_txt + 0,
                                           mod_txt + DMODEL, xm);
  ln_mod_kernel<<<LIMG, b256, 0, stream>>>(res_img, mod_img + 0,
                                           mod_img + DMODEL,
                                           xm + (size_t)LTXT * DMODEL);

  // QKV GEMMs
  gemm_bf16_kernel<<<dim3(6144 / 128, LTXT / 128), b256, 0, stream>>>(
      xm, qkvT_txt, txt_qkv_b, DMODEL, 0, qkvf, 3 * DMODEL, nullptr, 0, nullptr,
      nullptr, 0);
  gemm_bf16_kernel<<<dim3(6144 / 128, LIMG / 128), b256, 0, stream>>>(
      xm + (size_t)LTXT * DMODEL, qkvT_img, img_qkv_b, DMODEL, 0,
      qkvf + (size_t)LTXT * 3 * DMODEL, 3 * DMODEL, nullptr, 0, nullptr,
      nullptr, 0);

  // RMS + RoPE + pack
  qk_rope_pack_kernel<<<dim3(LSEQ, NH), 128, 0, stream>>>(
      qkvf, pe, img_qnorm, img_knorm, txt_qnorm, txt_knorm, qbuf, kbuf, vtb);

  // attention
  attn_kernel<<<dim3(LSEQ / 128, NH), 128, 0, stream>>>(qbuf, kbuf, vtb, attnb);

  // output projection, gated residual
  gemm_bf16_kernel<<<dim3(DMODEL / 128, LTXT / 128), b256, 0, stream>>>(
      attnb, projT_txt, txt_proj_b, DMODEL, 2, nullptr, 0, nullptr, 0,
      mod_txt + 2 * DMODEL, res_txt, DMODEL);
  gemm_bf16_kernel<<<dim3(DMODEL / 128, LIMG / 128), b256, 0, stream>>>(
      attnb + (size_t)LTXT * DMODEL, projT_img, img_proj_b, DMODEL, 2, nullptr,
      0, nullptr, 0, mod_img + 2 * DMODEL, res_img, DMODEL);

  // LN2 + modulate
  ln_mod_kernel<<<LTXT, b256, 0, stream>>>(res_txt, mod_txt + 3 * DMODEL,
                                           mod_txt + 4 * DMODEL, xm);
  ln_mod_kernel<<<LIMG, b256, 0, stream>>>(res_img, mod_img + 3 * DMODEL,
                                           mod_img + 4 * DMODEL,
                                           xm + (size_t)LTXT * DMODEL);

  // MLP0 (gelu, bf16 out)
  gemm_bf16_kernel<<<dim3(DMLP / 128, LTXT / 128), b256, 0, stream>>>(
      xm, mlp0T_txt, txt_mlp0_b, DMODEL, 1, nullptr, 0, hid, DMLP, nullptr,
      nullptr, 0);
  gemm_bf16_kernel<<<dim3(DMLP / 128, LIMG / 128), b256, 0, stream>>>(
      xm + (size_t)LTXT * DMODEL, mlp0T_img, img_mlp0_b, DMODEL, 1, nullptr, 0,
      hid + (size_t)LTXT * DMLP, DMLP, nullptr, nullptr, 0);

  // MLP1, gated residual
  gemm_bf16_kernel<<<dim3(DMODEL / 128, LTXT / 128), b256, 0, stream>>>(
      hid, mlp1T_txt, txt_mlp1_b, DMLP, 2, nullptr, 0, nullptr, 0,
      mod_txt + 5 * DMODEL, res_txt, DMODEL);
  gemm_bf16_kernel<<<dim3(DMODEL / 128, LIMG / 128), b256, 0, stream>>>(
      hid + (size_t)LTXT * DMLP, mlp1T_img, img_mlp1_b, DMLP, 2, nullptr, 0,
      nullptr, 0, mod_img + 5 * DMODEL, res_img, DMODEL);
}
